// _MessagePassingLayer_39994735460757
// MI455X (gfx1250) — matrix operand order repaired and checked by reading, not compiled, not run on hardware
//
#include <hip/hip_runtime.h>
#include <hip/hip_bf16.h>
#include <math.h>

// Problem constants (match reference)
#define NXC   4096
#define BBC   8
#define DLC   128     // D_LATENT
#define DHC   256     // D_HIDDEN
#define DTC   16      // D_TIME
#define KKC   3       // K
#define SSC   7       // 2K+1
#define KP1C  288     // padded first-layer input dim (275 / 272 -> 288 = 9*32)
#define STIN  296     // LDS row stride (halves) for input tiles (8-half aligned)
#define STH   264     // LDS row stride (halves) for hidden tiles

typedef _Float16 f16;
typedef __attribute__((ext_vector_type(16))) _Float16 v16h;
typedef __attribute__((ext_vector_type(8)))  _Float16 v8h;
typedef __attribute__((ext_vector_type(4)))  _Float16 v4h;
typedef __attribute__((ext_vector_type(8)))  float    v8f;

// Branch-free erf (Abramowitz-Stegun 7.1.26, max abs err ~1.5e-7).
// No EXEC divergence: 5 FMAs + v_rcp_f32 + v_exp_f32.
__device__ __forceinline__ float erf_bf(float x) {
    const float ax = fabsf(x);
    const float t  = __builtin_amdgcn_rcpf(1.0f + 0.3275911f * ax);
    float p = 1.061405429f;
    p = fmaf(p, t, -1.453152027f);
    p = fmaf(p, t,  1.421413741f);
    p = fmaf(p, t, -0.284496736f);
    p = fmaf(p, t,  0.254829592f);
    const float y = 1.0f - p * t * __expf(-ax * ax);
    return copysignf(y, x);
}

__device__ __forceinline__ float gelu_f(float v) {
    return 0.5f * v * (1.0f + erf_bf(v * 0.70710678118654752f));
}

__device__ __forceinline__ float sigmoid_f(float v) {
    return __builtin_amdgcn_rcpf(1.0f + __expf(-v));
}

// Issue global_prefetch for a weight block (spread across the 256 threads).
__device__ __forceinline__ void prefetch_block(const void* p, int bytes, int tid) {
    const char* c = (const char*)p;
    for (int off = tid * 128; off < bytes; off += 256 * 128)
        __builtin_prefetch(c + off, 0, 1);
}

// A-fragment (16-bit, 16x32) from LDS row-major tile.
// lanes 0-15: row m0+lane, K = k0+0..15
// lanes 16-31: row m0+lane-16, K = k0+16..31
__device__ __forceinline__ v16h lds_a_frag(const f16* base, int stride, int m0, int k0, int lane) {
    const int hi  = lane >> 4;
    const int row = m0 + (lane & 15);
    const f16* p = base + row * stride + k0 + hi * 16;
    v8h c0 = *(const v8h*)(p);
    v8h c1 = *(const v8h*)(p + 8);
    union { v16h v; v8h h[2]; } u;
    u.h[0] = c0; u.h[1] = c1;
    return u.v;
}

// B-fragment (16-bit, 32x16) from global W^T [N][kp] row-major (f16, zero padded).
// lanes 0-15: column n0+lane, K = k0+0..15 ; lanes 16-31: column n0+lane-16, K = k0+16..31
__device__ __forceinline__ v16h ldg_b_frag(const f16* wT, int kp, int n0, int k0, int lane) {
    const int n = n0 + (lane & 15);
    const int k = k0 + ((lane >> 4) << 4);
    return *(const v16h*)(wT + n * kp + k);
}

// Store a 16x16 f32 accumulator tile to LDS as f16 after bias + GELU.
// C layout: VGPR v, lanes 0-15 -> (M=v, N=lane), lanes 16-31 -> (M=v+8, N=lane-16)
__device__ __forceinline__ void store_tile_gelu(f16* out, int stride, int m0, int n0,
                                                v8f c, const float* bias, int lane) {
    const int hi  = lane >> 4;
    const int col = n0 + (lane & 15);
    const float bv = bias[col];
#pragma unroll
    for (int v = 0; v < 8; ++v) {
        const int row = m0 + v + hi * 8;
        out[row * stride + col] = (f16)gelu_f(c[v] + bv);
    }
}

#define WMMA_F16(a, b, c) \
    __builtin_amdgcn_wmma_f32_16x16x32_f16(false, (a), false, (b), (short)0, (c), false, false)

// ---------------------------------------------------------------------------
// Weight prep: fp32 [din][dout] row-major  ->  f16 [npad][kp] = W^T zero-padded
// ---------------------------------------------------------------------------
__global__ void prep_weights(const float* __restrict__ src, f16* __restrict__ dst,
                             int din, int dout, int kp, int npad) {
    int idx = blockIdx.x * blockDim.x + threadIdx.x;
    int total = npad * kp;
    if (idx >= total) return;
    int n = idx / kp;
    int k = idx - n * kp;
    f16 v = (f16)0.0f;
    if (n < dout && k < din) v = (f16)src[k * dout + n];
    dst[idx] = v;
}

// ---------------------------------------------------------------------------
// Edge kernel: one WG = 16 nodes x 7 neighbors = 112 edge rows.
// Computes gate MLP + msg MLP with WMMA, aggregates gate*msg over S -> agg.
// ---------------------------------------------------------------------------
__global__ __launch_bounds__(256) void edge_kernel(
    const float* __restrict__ h, const float* __restrict__ x,
    const float* __restrict__ tau, const float* __restrict__ u0,
    const f16* __restrict__ msg_w1T, const float* __restrict__ msg_b1,
    const f16* __restrict__ msg_w2T, const float* __restrict__ msg_b2,
    const f16* __restrict__ msg_w3T, const float* __restrict__ msg_b3,
    const f16* __restrict__ gate_w1T, const float* __restrict__ gate_b1,
    const f16* __restrict__ gate_w2T, const float* __restrict__ gate_b2,
    float* __restrict__ agg)
{
    extern __shared__ char smem[];
    f16*   sIn   = (f16*)smem;                                  // 112 x STIN halves
    f16*   sA    = (f16*)(smem + 112 * STIN * 2);               // 112 x STH halves
    f16*   sB    = sA + 112 * STH;                              // 112 x STH halves
    float* sGate = (float*)(sB + 112 * STH);                    // 112 floats

    const int tid  = threadIdx.x;
    const int lane = tid & 31;
    const int wave = tid >> 5;
    const int i0   = blockIdx.x * 16;
    const int b    = blockIdx.y;

    const float* hb  = h  + (size_t)b * NXC * DLC;
    const float* xb  = x  + (size_t)b * NXC;
    const float* u0b = u0 + (size_t)b * NXC;
    const float* tb  = tau + (size_t)b * DTC;

    // Warm GL2 with the first-layer weights while we build the input tile.
    prefetch_block(gate_w1T, 256 * KP1C * 2, tid);

    // ---- Build msg_in tile [112][288] (f16, zero-padded) ----
    // Vectorized part: cols 0..255 (h_e | h_j) via float4 loads -> 4xf16 stores.
    for (int e = tid; e < 112 * 64; e += 256) {
        const int r   = e >> 6;          // edge row
        const int col = (e & 63) * 4;    // 0..252 step 4
        const int j   = r >> 4;
        const int li  = r & 15;
        const int i   = i0 + li;
        int jn = i + j - KKC;
        jn = jn < 0 ? 0 : (jn >= NXC ? NXC - 1 : jn);
        const float* src = (col < 128) ? (hb + i * DLC + col)
                                       : (hb + jn * DLC + (col - 128));
        const float4 v = *(const float4*)src;
        v4h o; o.x = (f16)v.x; o.y = (f16)v.y; o.z = (f16)v.z; o.w = (f16)v.w;
        *(v4h*)(&sIn[r * STIN + col]) = o;
    }
    // Scalar tail: cols 256..287 (rel | tau | du0 | |du0| | zero-pad).
    for (int e = tid; e < 112 * 32; e += 256) {
        const int r   = e >> 5;
        const int col = 256 + (e & 31);
        const int j   = r >> 4;
        const int li  = r & 15;
        const int i   = i0 + li;
        int jn = i + j - KKC;
        jn = jn < 0 ? 0 : (jn >= NXC ? NXC - 1 : jn);
        float v;
        if (col == 256)       v = xb[jn] - xb[i];
        else if (col < 273)   v = tb[col - 257];
        else if (col == 273)  v = u0b[i] - u0b[jn];
        else if (col == 274)  v = fabsf(u0b[i] - u0b[jn]);
        else                  v = 0.0f;
        sIn[r * STIN + col] = (f16)v;
    }
    __syncthreads();

    // ---- Gate layer 1: [112x288] @ [288x256] -> gelu -> sB ----
    prefetch_block(msg_w1T, 256 * KP1C * 2, tid);
#pragma unroll
    for (int nn = 0; nn < 2; ++nn) {
        const int n0 = (wave * 2 + nn) * 16;
        for (int m = 0; m < 7; ++m) {
            v8f c = {};
#pragma unroll
            for (int kt = 0; kt < 9; ++kt) {
                v16h a = lds_a_frag(sIn, STIN, m * 16, kt * 32, lane);
                v16h bfrag = ldg_b_frag(gate_w1T, KP1C, n0, kt * 32, lane);
                c = WMMA_F16(a, bfrag, c);
            }
            store_tile_gelu(sB, STH, m * 16, n0, c, gate_b1, lane);
        }
    }
    __syncthreads();

    // ---- Gate layer 2: [112x256] @ [256x16(pad of 1)] -> sigmoid -> sGate ----
    if (wave < 7) {
        const int m = wave;
        v8f c = {};
#pragma unroll
        for (int kt = 0; kt < 8; ++kt) {
            v16h a = lds_a_frag(sB, STH, m * 16, kt * 32, lane);
            v16h bfrag = ldg_b_frag(gate_w2T, 256, 0, kt * 32, lane);
            c = WMMA_F16(a, bfrag, c);
        }
        if ((lane & 15) == 0) {
            const int hi = lane >> 4;
            const float gb = gate_b2[0];
#pragma unroll
            for (int v = 0; v < 8; ++v) {
                const int row = m * 16 + v + hi * 8;
                sGate[row] = sigmoid_f(c[v] + gb);
            }
        }
    }
    __syncthreads();

    // ---- Msg layer 1: sIn @ msg_w1 -> gelu -> sA ----
    prefetch_block(msg_w2T, 256 * 256 * 2, tid);
#pragma unroll
    for (int nn = 0; nn < 2; ++nn) {
        const int n0 = (wave * 2 + nn) * 16;
        for (int m = 0; m < 7; ++m) {
            v8f c = {};
#pragma unroll
            for (int kt = 0; kt < 9; ++kt) {
                v16h a = lds_a_frag(sIn, STIN, m * 16, kt * 32, lane);
                v16h bfrag = ldg_b_frag(msg_w1T, KP1C, n0, kt * 32, lane);
                c = WMMA_F16(a, bfrag, c);
            }
            store_tile_gelu(sA, STH, m * 16, n0, c, msg_b1, lane);
        }
    }
    __syncthreads();

    // ---- Msg layer 2: sA @ msg_w2 -> gelu -> sB (overwrites gate hidden) ----
    prefetch_block(msg_w3T, 128 * 256 * 2, tid);
#pragma unroll
    for (int nn = 0; nn < 2; ++nn) {
        const int n0 = (wave * 2 + nn) * 16;
        for (int m = 0; m < 7; ++m) {
            v8f c = {};
#pragma unroll
            for (int kt = 0; kt < 8; ++kt) {
                v16h a = lds_a_frag(sA, STH, m * 16, kt * 32, lane);
                v16h bfrag = ldg_b_frag(msg_w2T, 256, n0, kt * 32, lane);
                c = WMMA_F16(a, bfrag, c);
            }
            store_tile_gelu(sB, STH, m * 16, n0, c, msg_b2, lane);
        }
    }
    __syncthreads();

    // ---- Msg layer 3 + gated aggregation over S ----
    {
        const int n0  = wave * 16;                 // 8 waves x 16 cols = 128
        const int hi  = lane >> 4;
        const int col = n0 + (lane & 15);
        const float b3 = msg_b3[col];
        float acc[8];
#pragma unroll
        for (int v = 0; v < 8; ++v) acc[v] = 0.0f;

        for (int j = 0; j < 7; ++j) {
            v8f c = {};
#pragma unroll
            for (int kt = 0; kt < 8; ++kt) {
                v16h a = lds_a_frag(sB, STH, j * 16, kt * 32, lane);
                v16h bfrag = ldg_b_frag(msg_w3T, 256, n0, kt * 32, lane);
                c = WMMA_F16(a, bfrag, c);
            }
#pragma unroll
            for (int v = 0; v < 8; ++v) {
                const int row = j * 16 + v + hi * 8;
                acc[v] += sGate[row] * (c[v] + b3);
            }
        }
#pragma unroll
        for (int v = 0; v < 8; ++v) {
            const int li = v + hi * 8;
            agg[((size_t)b * NXC + (i0 + li)) * DLC + col] = acc[v];
        }
    }
}

// ---------------------------------------------------------------------------
// Node kernel: upd MLP + local linear, fused GELU output.
// ---------------------------------------------------------------------------
__global__ __launch_bounds__(256) void node_kernel(
    const float* __restrict__ h, const float* __restrict__ tau,
    const float* __restrict__ agg,
    const f16* __restrict__ upd_w1T, const float* __restrict__ upd_b1,
    const f16* __restrict__ upd_w2T, const float* __restrict__ upd_b2,
    const f16* __restrict__ upd_w3T, const float* __restrict__ upd_b3,
    const f16* __restrict__ WwT,     const float* __restrict__ Wb,
    float* __restrict__ out)
{
    extern __shared__ char smem[];
    f16* sIn = (f16*)smem;              // 16 x STIN halves ([h | agg | tau | 0])
    f16* sA  = sIn + 16 * STIN;         // 16 x STH
    f16* sB  = sA + 16 * STH;           // 16 x STH

    const int tid  = threadIdx.x;
    const int lane = tid & 31;
    const int wave = tid >> 5;
    const int i0   = blockIdx.x * 16;
    const int b    = blockIdx.y;

    // ---- Build upd_in tile [16][288] ----
    for (int e = tid; e < 16 * 64; e += 256) {
        const int r   = e >> 6;
        const int col = (e & 63) * 4;
        const int i   = i0 + r;
        const float* src = (col < 128) ? (h + ((size_t)b * NXC + i) * DLC + col)
                                       : (agg + ((size_t)b * NXC + i) * DLC + (col - 128));
        const float4 v = *(const float4*)src;
        v4h o; o.x = (f16)v.x; o.y = (f16)v.y; o.z = (f16)v.z; o.w = (f16)v.w;
        *(v4h*)(&sIn[r * STIN + col]) = o;
    }
    for (int e = tid; e < 16 * 32; e += 256) {
        const int r   = e >> 5;
        const int col = 256 + (e & 31);
        float v = (col < 272) ? tau[b * DTC + (col - 256)] : 0.0f;
        sIn[r * STIN + col] = (f16)v;
    }
    __syncthreads();

    // ---- upd layer 1: [16x288] @ [288x256] -> gelu -> sA ----
    prefetch_block(upd_w2T, 256 * 256 * 2, tid);
#pragma unroll
    for (int nn = 0; nn < 2; ++nn) {
        const int n0 = (wave * 2 + nn) * 16;
        v8f c = {};
#pragma unroll
        for (int kt = 0; kt < 9; ++kt) {
            v16h a = lds_a_frag(sIn, STIN, 0, kt * 32, lane);
            v16h bfrag = ldg_b_frag(upd_w1T, KP1C, n0, kt * 32, lane);
            c = WMMA_F16(a, bfrag, c);
        }
        store_tile_gelu(sA, STH, 0, n0, c, upd_b1, lane);
    }
    __syncthreads();

    // ---- upd layer 2: [16x256] @ [256x256] -> gelu -> sB ----
    prefetch_block(upd_w3T, 128 * 256 * 2, tid);
#pragma unroll
    for (int nn = 0; nn < 2; ++nn) {
        const int n0 = (wave * 2 + nn) * 16;
        v8f c = {};
#pragma unroll
        for (int kt = 0; kt < 8; ++kt) {
            v16h a = lds_a_frag(sA, STH, 0, kt * 32, lane);
            v16h bfrag = ldg_b_frag(upd_w2T, 256, n0, kt * 32, lane);
            c = WMMA_F16(a, bfrag, c);
        }
        store_tile_gelu(sB, STH, 0, n0, c, upd_b2, lane);
    }
    __syncthreads();

    // ---- upd layer 3 (sB @ upd_w3) + local (h @ W_w), fused output GELU ----
    {
        const int n0 = wave * 16;                  // 8 waves x 16 cols = 128
        v8f c1 = {};
#pragma unroll
        for (int kt = 0; kt < 8; ++kt) {
            v16h a = lds_a_frag(sB, STH, 0, kt * 32, lane);
            v16h bfrag = ldg_b_frag(upd_w3T, 256, n0, kt * 32, lane);
            c1 = WMMA_F16(a, bfrag, c1);
        }
        v8f c2 = {};
#pragma unroll
        for (int kt = 0; kt < 4; ++kt) {
            v16h a = lds_a_frag(sIn, STIN, 0, kt * 32, lane);  // cols 0..127 == h (f16)
            v16h bfrag = ldg_b_frag(WwT, 128, n0, kt * 32, lane);
            c2 = WMMA_F16(a, bfrag, c2);
        }
        const int hi  = lane >> 4;
        const int col = n0 + (lane & 15);
        const float bb = upd_b3[col] + Wb[col];
#pragma unroll
        for (int v = 0; v < 8; ++v) {
            const int i = i0 + v + hi * 8;
            out[((size_t)b * NXC + i) * DLC + col] = gelu_f(c1[v] + c2[v] + bb);
        }
    }
}

// ---------------------------------------------------------------------------
// Host launcher
// ---------------------------------------------------------------------------
extern "C" void kernel_launch(void* const* d_in, const int* in_sizes, int n_in,
                              void* d_out, int out_size, void* d_ws, size_t ws_size,
                              hipStream_t stream) {
    const float* h       = (const float*)d_in[0];
    const float* x       = (const float*)d_in[1];
    const float* tau     = (const float*)d_in[2];
    const float* u0      = (const float*)d_in[3];
    const float* msg_w1  = (const float*)d_in[4];
    const float* msg_b1  = (const float*)d_in[5];
    const float* msg_w2  = (const float*)d_in[6];
    const float* msg_b2  = (const float*)d_in[7];
    const float* msg_w3  = (const float*)d_in[8];
    const float* msg_b3  = (const float*)d_in[9];
    const float* gate_w1 = (const float*)d_in[10];
    const float* gate_b1 = (const float*)d_in[11];
    const float* gate_w2 = (const float*)d_in[12];
    const float* gate_b2 = (const float*)d_in[13];
    const float* upd_w1  = (const float*)d_in[14];
    const float* upd_b1  = (const float*)d_in[15];
    const float* upd_w2  = (const float*)d_in[16];
    const float* upd_b2  = (const float*)d_in[17];
    const float* upd_w3  = (const float*)d_in[18];
    const float* upd_b3  = (const float*)d_in[19];
    const float* W_w     = (const float*)d_in[20];
    const float* W_b     = (const float*)d_in[21];
    float* out = (float*)d_out;

    char* ws = (char*)d_ws;
    // f16 transposed/padded weights in workspace
    f16* msg_w1T  = (f16*)(ws + 0);        // 256 x 288
    f16* msg_w2T  = (f16*)(ws + 147456);   // 256 x 256
    f16* msg_w3T  = (f16*)(ws + 278528);   // 128 x 256
    f16* gate_w1T = (f16*)(ws + 344064);   // 256 x 288
    f16* gate_w2T = (f16*)(ws + 491520);   // 16  x 256 (only row 0 nonzero)
    f16* upd_w1T  = (f16*)(ws + 499712);   // 256 x 288
    f16* upd_w2T  = (f16*)(ws + 647168);   // 256 x 256
    f16* upd_w3T  = (f16*)(ws + 778240);   // 128 x 256
    f16* WwT      = (f16*)(ws + 843776);   // 128 x 128
    float* agg    = (float*)(ws + 1048576); // B*NX*128 fp32 = 16 MB

    auto launch_prep = [&](const float* src, f16* dst, int din, int dout, int kp, int npad) {
        int total = npad * kp;
        prep_weights<<<dim3((total + 255) / 256), dim3(256), 0, stream>>>(src, dst, din, dout, kp, npad);
    };
    launch_prep(msg_w1,  msg_w1T,  275, 256, KP1C, 256);
    launch_prep(msg_w2,  msg_w2T,  256, 256, 256,  256);
    launch_prep(msg_w3,  msg_w3T,  256, 128, 256,  128);
    launch_prep(gate_w1, gate_w1T, 275, 256, KP1C, 256);
    launch_prep(gate_w2, gate_w2T, 256, 1,   256,  16);
    launch_prep(upd_w1,  upd_w1T,  272, 256, KP1C, 256);
    launch_prep(upd_w2,  upd_w2T,  256, 256, 256,  256);
    launch_prep(upd_w3,  upd_w3T,  256, 128, 256,  128);
    launch_prep(W_w,     WwT,      128, 128, 128,  128);

    constexpr int EDGE_LDS = (112 * STIN + 2 * 112 * STH) * 2 + 112 * 4; // 185024 B (CDNA5: 320KB/WGP)
    constexpr int NODE_LDS = (16 * STIN + 2 * 16 * STH) * 2;             // 26368 B

    edge_kernel<<<dim3(NXC / 16, BBC), dim3(256), EDGE_LDS, stream>>>(
        h, x, tau, u0,
        msg_w1T, msg_b1, msg_w2T, msg_b2, msg_w3T, msg_b3,
        gate_w1T, gate_b1, gate_w2T, gate_b2, agg);

    node_kernel<<<dim3(NXC / 16, BBC), dim3(256), NODE_LDS, stream>>>(
        h, tau, agg,
        upd_w1T, upd_b1, upd_w2T, upd_b2, upd_w3T, upd_b3,
        WwT, W_b, out);
}
